// TransferModel_41171556499584
// MI455X (gfx1250) — matrix operand order repaired and checked by reading, not compiled, not run on hardware
//
#include <hip/hip_runtime.h>
#include <hip/hip_bf16.h>

// ---------------------------------------------------------------------------
// MI455X / gfx1250 implementation. wave32, WMMA 16x16x32 bf16 -> f32.
// Heavy math (conv3d implicit GEMM + all linear layers) runs on the matrix
// pipe; fp32 operands are converted to bf16 while staging LDS tiles.
// LDS tile layouts chosen so each lane's WMMA fragment is contiguous:
//   A: [M][K] (row stride 72 elem, 16B aligned)  -> 2x ds_load_b128 / frag
//   B: transposed [N][K]                          -> 2x ds_load_b128 / frag
// ---------------------------------------------------------------------------

typedef __bf16 bf16;
typedef __attribute__((ext_vector_type(16))) __bf16 v16bf;
typedef __attribute__((ext_vector_type(8)))  __bf16 v8bf;
typedef __attribute__((ext_vector_type(8)))  float  v8f;

// ---------------- model dims ----------------
#define NB   4
#define NT   64
#define ND   512
#define NHD  128
#define NNH  4
#define CONV_M   12544            // B*T*7*7
#define CONV_N   512
#define CONV_K   20736            // 768*27
#define FEAT_PER_SCALE (4*768*64*49)

// ---------------- input indices (setup_inputs dict order) ----------------
enum {
  IN_FEATS = 0,
  IN_CONV3D_W, IN_CONV3D_B, IN_BN1_G, IN_BN1_B, IN_BN1_M, IN_BN1_V,
  IN_WQ, IN_BQ, IN_WK, IN_BK, IN_DW_W, IN_DW_B, IN_PW_W, IN_PW_B,
  IN_FC_W, IN_FC_B, IN_FCLN_G, IN_FCLN_B,
  IN_C1_W, IN_C3_W, IN_C5_W, IN_CALL_W,
  IN_CONV2_W, IN_CONV2_B, IN_BN2_G, IN_BN2_B, IN_BN2_M, IN_BN2_V,
  IN_PROJ_W, IN_PROJ_B, IN_LN1_G, IN_LN1_B,
  IN_ATTN_IN_W, IN_ATTN_IN_B, IN_ATTN_OUT_W, IN_ATTN_OUT_B,
  IN_FF1_W, IN_FF1_B, IN_FF2_W, IN_FF2_B,
  IN_ELN1_G, IN_ELN1_B, IN_ELN2_G, IN_ELN2_B, IN_ENORM_G, IN_ENORM_B,
  IN_PRED1_W, IN_PRED1_B, IN_PREDLN_G, IN_PREDLN_B,
  IN_PRED2_W, IN_PRED2_B, IN_PRED3_W, IN_PRED3_B
};

// ---------------- output layout (floats, concat in return order) ----------
#define OUT_H    0
#define OUT_XMAT 256
#define OUT_O    (256 + 245760)
#define OUT_M1   (OUT_O + 131072)
#define OUT_M2   (OUT_M1 + 196608)

// ---------------- workspace layout (float offsets) ----------------
#define WS_CONVOUT 0                      // [12544][512]
#define WS_H       6422528                // [4][512][64]
#define WS_XS      6553600                // [4][64][512]
#define WS_Q       6684672
#define WS_K       6815744
#define WS_RAW     6946816                // [4][4][64][64]
#define WS_T1      7012352                // [4][64][512]
#define WS_TPW     7143424                // [4][64][512]
#define WS_HATF    7274496                // [4][64][64]
#define WS_CAT     7290880                // [4][45][4096]
#define WS_ALLC    8028160                // [4][15] (padded)
#define WS_X15     8028224                // [4][15][4096]
#define WS_XT      8273984                // [4][64][64][32] NHWC
#define WS_XP      8798272
#define WS_X0      8929344
#define WS_QKV     9060416                // [4][64][1536]
#define WS_RAW2    9453632                // [4][4][64][64]
#define WS_OBUF    9519168
#define WS_X1T     9650240
#define WS_X1      9781312
#define WS_F1      9912384
#define WS_X2T     10043456
#define WS_X2      10174528
#define WS_P1T     10305600
#define WS_P1      10436672
#define WS_P2      10567744               // [4][64][256]

#define LDS_STRIDE 72                      // elems; 144B rows (16B multiple)

// ===========================================================================
// Fragment loads: contiguous K per lane (ISA 7.12.2 layouts), b128-friendly.
//   A lane (row m): elems 0..7 -> K = kb+8*half+e ; 8..15 -> K = kb+16+8*half+e
//   B lane (col n): elems 0..7 -> K = kb+8*half+e ; 8..15 -> K = kb+16+8*half+e
// ===========================================================================
__device__ __forceinline__ v16bf frag_a(const bf16* __restrict__ rowBase, int kb, int half)
{
  const v8bf lo = *(const v8bf*)(rowBase + kb + 8 * half);
  const v8bf hi = *(const v8bf*)(rowBase + kb + 16 + 8 * half);
  return __builtin_shufflevector(lo, hi, 0,1,2,3,4,5,6,7,8,9,10,11,12,13,14,15);
}
__device__ __forceinline__ v16bf frag_b(const bf16* __restrict__ colBase, int kb, int half)
{
  const v8bf lo = *(const v8bf*)(colBase + kb + 8 * half);
  const v8bf hi = *(const v8bf*)(colBase + kb + 16 + 8 * half);
  return __builtin_shufflevector(lo, hi, 0,1,2,3,4,5,6,7,8,9,10,11,12,13,14,15);
}

// ===========================================================================
// Implicit-GEMM dilated conv3d: C[m][o] += A[m][k] * W[o*K + k]
//   m = ((b*64+t)*7+y)*7+x ; k = ((ci*3+kd)*3+kh)*3+kw
//   t_in = t + 3*kd - 3 ; y_in = y + kh - 1 ; x_in = x + kw - 1 (zero pad)
// Block: 256 thr = 8 waves (4 M x 2 N) -> 64 x 32 tile of C; K slab = 64.
// ===========================================================================
__global__ void k_conv3d(const float* __restrict__ F, const float* __restrict__ W,
                         float* __restrict__ Cout)
{
  __shared__ alignas(16) bf16 As[64][LDS_STRIDE];
  __shared__ alignas(16) bf16 Bst[32][LDS_STRIDE];   // transposed: [o][k]
  const int m0 = blockIdx.x * 64, n0 = blockIdx.y * 32;
  const int tid = threadIdx.x, wave = tid >> 5, lane = tid & 31;
  const int wm = wave >> 1, wn = wave & 1;
  const int half = lane >> 4, mr = lane & 15;

  // per-thread fixed A-row coordinates (rows r0 + 8*j), K-col c fixed
  const int c  = tid & 31;
  const int r0 = tid >> 5;
  int rt[8], ry[8], rx[8];
  long long baseA[8];
#pragma unroll
  for (int j = 0; j < 8; ++j) {
    const int gm = m0 + r0 + j * 8;
    rx[j] = gm % 7; ry[j] = (gm / 7) % 7; rt[j] = (gm / 49) % 64;
    const int rb = gm / 3136;
    baseA[j] = (((long long)rb * 768) * 64 + rt[j]) * 49 + ry[j] * 7 + rx[j];
  }

  v8f acc = {0.f,0.f,0.f,0.f,0.f,0.f,0.f,0.f};
  for (int k0 = 0; k0 < CONV_K; k0 += 64) {
    __syncthreads();
    // ---- stage A 64x64 (branchless clamped fp32 gather -> bf16) ----
#pragma unroll
    for (int hb = 0; hb < 2; ++hb) {
      const int gk = k0 + hb * 32 + c;
      const int kw = gk % 3, kh = (gk / 3) % 3, kd = (gk / 9) % 3, ci = gk / 27;
      const int dt = 3 * kd - 3, dy = kh - 1, dx = kw - 1;
      const long long koff = (long long)ci * 3136 + dt * 49 + dy * 7 + dx;
#pragma unroll
      for (int j = 0; j < 8; ++j) {
        const bool ok = ((unsigned)(rt[j] + dt) < 64u) &
                        ((unsigned)(ry[j] + dy) < 7u) &
                        ((unsigned)(rx[j] + dx) < 7u);
        const long long a = ok ? (baseA[j] + koff) : 0;
        const float v = F[a];
        As[r0 + j * 8][hb * 32 + c] = (bf16)(ok ? v : 0.f);
      }
    }
    // ---- stage B transposed 32x64: Bst[o][k] = W[(n0+o)*K + k0+k] ----
    for (int i = tid; i < 32 * 64; i += 256) {
      const int n = i >> 6, k = i & 63;
      Bst[n][k] = (bf16)W[(long long)(n0 + n) * CONV_K + (k0 + k)];
    }
    if (k0 + 64 < CONV_K)
      __builtin_prefetch(&W[(long long)(n0 + (tid >> 3)) * CONV_K + k0 + 64 + (tid & 7) * 8], 0, 1);
    __syncthreads();
    const bf16* pa = &As[wm * 16 + mr][0];
    const bf16* pb = &Bst[wn * 16 + mr][0];
    acc = __builtin_amdgcn_wmma_f32_16x16x32_bf16(false, frag_a(pa, 0, half),
                                                  false, frag_b(pb, 0, half),
                                                  (short)0, acc, false, false);
    acc = __builtin_amdgcn_wmma_f32_16x16x32_bf16(false, frag_a(pa, 32, half),
                                                  false, frag_b(pb, 32, half),
                                                  (short)0, acc, false, false);
  }
  const int gn = n0 + wn * 16 + mr;
#pragma unroll
  for (int r = 0; r < 8; ++r) {
    const int gm = m0 + wm * 16 + half * 8 + r;
    Cout[(long long)gm * 512 + gn] = acc[r];
  }
}

// ===========================================================================
// Generic tiled GEMM:  C = alpha*(A@B) + bias + Res  (optional ReLU)
// BT=true : B given as W[N][K] row-major (x @ W.T);  BT=false: B is [K][N].
// Batched via blockIdx.z with split strides: base = (z/zdiv)*s1 + (z%zdiv)*s2.
// ===========================================================================
template <bool BT>
__global__ void k_gemm(const float* __restrict__ A, const float* __restrict__ B,
                       const float* __restrict__ bias, const float* __restrict__ Res,
                       float* __restrict__ C,
                       int M, int N, int K, int lda, int ldb, int ldc,
                       int zdiv, long long aZ1, long long aZ2,
                       long long bZ1, long long bZ2, long long cZ1, long long cZ2,
                       float alpha, int flags)
{
  const int z = blockIdx.z;
  A += (z / zdiv) * aZ1 + (z % zdiv) * aZ2;
  B += (z / zdiv) * bZ1 + (z % zdiv) * bZ2;
  C += (z / zdiv) * cZ1 + (z % zdiv) * cZ2;
  if (Res) Res += (z / zdiv) * cZ1 + (z % zdiv) * cZ2;

  __shared__ alignas(16) bf16 As[64][LDS_STRIDE];
  __shared__ alignas(16) bf16 Bst[32][LDS_STRIDE];   // transposed: [n][k]
  const int m0 = blockIdx.x * 64, n0 = blockIdx.y * 32;
  const int tid = threadIdx.x, wave = tid >> 5, lane = tid & 31;
  const int wm = wave >> 1, wn = wave & 1;
  const int half = lane >> 4, mr = lane & 15;

  v8f acc = {0.f,0.f,0.f,0.f,0.f,0.f,0.f,0.f};
  for (int k0 = 0; k0 < K; k0 += 64) {
    __syncthreads();
    for (int i = tid; i < 64 * 64; i += 256) {
      const int r = i >> 6, cc = i & 63;
      const int gm = m0 + r, gk = k0 + cc;
      const bool ok = (gm < M) & (gk < K);
      const long long idx = ok ? ((long long)gm * lda + gk) : 0;
      const float v = A[idx];
      As[r][cc] = (bf16)(ok ? v : 0.f);
    }
    for (int i = tid; i < 32 * 64; i += 256) {
      const int n = i >> 6, k = i & 63;
      const int gk = k0 + k, gn = n0 + n;
      const bool ok = (gk < K) & (gn < N);
      const long long idx = !ok ? 0
                          : (BT ? ((long long)gn * ldb + gk) : ((long long)gk * ldb + gn));
      const float v = B[idx];
      Bst[n][k] = (bf16)(ok ? v : 0.f);
    }
    __syncthreads();
    const bf16* pa = &As[wm * 16 + mr][0];
    const bf16* pb = &Bst[wn * 16 + mr][0];
    acc = __builtin_amdgcn_wmma_f32_16x16x32_bf16(false, frag_a(pa, 0, half),
                                                  false, frag_b(pb, 0, half),
                                                  (short)0, acc, false, false);
    acc = __builtin_amdgcn_wmma_f32_16x16x32_bf16(false, frag_a(pa, 32, half),
                                                  false, frag_b(pb, 32, half),
                                                  (short)0, acc, false, false);
  }
  const int gn = n0 + wn * 16 + mr;
#pragma unroll
  for (int r = 0; r < 8; ++r) {
    const int gm = m0 + wm * 16 + half * 8 + r;
    if (gm < M && gn < N) {
      float v = alpha * acc[r];
      if (bias) v += bias[gn];
      if (Res)  v += Res[(long long)gm * ldc + gn];
      if (flags & 1) v = fmaxf(v, 0.f);
      C[(long long)gm * ldc + gn] = v;
    }
  }
}

// ===========================================================================
// BN1 + ReLU + spatial max over 7x7  -> h[b][o][t] and xs[b][t][o]
// ===========================================================================
__global__ void k_bnmax(const float* __restrict__ conv, const float* __restrict__ cb,
                        const float* __restrict__ g, const float* __restrict__ bt,
                        const float* __restrict__ mean, const float* __restrict__ var,
                        float* __restrict__ h, float* __restrict__ xs)
{
  const int idx = blockIdx.x * 256 + threadIdx.x;
  if (idx >= NB * ND * NT) return;
  const int o = idx % ND, t = (idx / ND) % NT, b = idx / (ND * NT);
  const float sc = g[o] * rsqrtf(var[o] + 1e-5f);
  const float sh = bt[o] - mean[o] * sc;
  const float bias = cb[o];
  const float* base = conv + ((long long)(b * NT + t) * 49) * 512 + o;
  float mx = 0.f;
  for (int p = 0; p < 49; ++p) {
    const float v = (base[(long long)p * 512] + bias) * sc + sh;
    mx = fmaxf(mx, fmaxf(v, 0.f));
  }
  h[((long long)b * ND + o) * NT + t]  = mx;
  xs[((long long)b * NT + t) * ND + o] = mx;
}

// depthwise conv1d k=5 pad=2 over t; output transposed [b][l][c]
__global__ void k_dwconv(const float* __restrict__ h, const float* __restrict__ w,
                         const float* __restrict__ bias, float* __restrict__ t1)
{
  const int idx = blockIdx.x * 256 + threadIdx.x;
  if (idx >= NB * ND * NT) return;
  const int t = idx % NT, c = (idx / NT) % ND, b = idx / (NT * ND);
  const float* row = h + ((long long)b * ND + c) * NT;
  float acc = bias[c];
#pragma unroll
  for (int j = 0; j < 5; ++j) {
    const int l = t + j - 2;
    if ((unsigned)l < (unsigned)NT) acc += w[c * 5 + j] * row[l];
  }
  t1[((long long)b * NT + t) * ND + c] = acc;
}

// ===========================================================================
// Dual softmax attention maps: s2 = softmax(raw/8, rows), s1 = colsoft*rowsoft
// Writes relu(s1+s2) into x_matrix channel, relu(s1)->m1, relu(s2)->m2.
// One block per (b,head), 64 threads.
// ===========================================================================
__global__ void k_attn_soft(const float* __restrict__ raw, float* __restrict__ xmat,
                            float* __restrict__ m1, float* __restrict__ m2, int scale)
{
  __shared__ float tile[64][65];
  __shared__ float rmax[64], rs1[64], rs2[64], cmax[64], cs[64];
  const int b = blockIdx.x >> 2, hh = blockIdx.x & 3;
  const int tid = threadIdx.x;
  const float* R = raw + (((long long)b * NNH + hh) << 12);
  for (int i = 0; i < 64; ++i) tile[i][tid] = R[(i << 6) + tid];
  __syncthreads();
  // row stats
  float mx = -3.4e38f;
  for (int j = 0; j < 64; ++j) mx = fmaxf(mx, tile[tid][j]);
  float s1 = 0.f, s2 = 0.f;
  for (int j = 0; j < 64; ++j) {
    const float d = tile[tid][j] - mx;
    s1 += __expf(d);
    s2 += __expf(d * 0.125f);
  }
  rmax[tid] = mx; rs1[tid] = s1; rs2[tid] = s2;
  // column stats
  float cm = -3.4e38f;
  for (int i = 0; i < 64; ++i) cm = fmaxf(cm, tile[i][tid]);
  float csum = 0.f;
  for (int i = 0; i < 64; ++i) csum += __expf(tile[i][tid] - cm);
  cmax[tid] = cm; cs[tid] = csum;
  __syncthreads();
  const long long xBase = ((long long)b * 15 + scale * 5 + hh) << 12;
  const long long mBase = ((long long)b * 12 + scale * 4 + hh) << 12;
  for (int i = 0; i < 64; ++i) {
    const float r  = tile[i][tid];
    const float v1 = (__expf(r - rmax[i]) / rs1[i]) * (__expf(r - cmax[tid]) / cs[tid]);
    const float v2 = __expf((r - rmax[i]) * 0.125f) / rs2[i];
    xmat[xBase + (i << 6) + tid] = fmaxf(v1 + v2, 0.f);
    m1[mBase + (i << 6) + tid]   = fmaxf(v1, 0.f);
    m2[mBase + (i << 6) + tid]   = fmaxf(v2, 0.f);
  }
}

// LayerNorm over 64 feats, write hat channel (scale*5+4) of x_matrix
__global__ void k_ln64_hat(const float* __restrict__ hatf, const float* __restrict__ g,
                           const float* __restrict__ bt, float* __restrict__ xmat, int scale)
{
  __shared__ float sm[64];
  const int row = blockIdx.x, b = row >> 6, l = row & 63, tid = threadIdx.x;
  const float v = hatf[(row << 6) + tid];
  sm[tid] = v; __syncthreads();
  for (int s = 32; s > 0; s >>= 1) { if (tid < s) sm[tid] += sm[tid + s]; __syncthreads(); }
  const float mean = sm[0] * (1.f / 64.f); __syncthreads();
  const float d = v - mean;
  sm[tid] = d * d; __syncthreads();
  for (int s = 32; s > 0; s >>= 1) { if (tid < s) sm[tid] += sm[tid + s]; __syncthreads(); }
  const float inv = rsqrtf(sm[0] * (1.f / 64.f) + 1e-5f);
  xmat[(((long long)b * 15 + scale * 5 + 4) << 12) + (l << 6) + tid] = d * inv * g[tid] + bt[tid];
}

// on-device sinusoidal PE (reference _make_pe)
__device__ __forceinline__ float pe_val(int t, int d)
{
  const int i = d >> 1;
  const float div = __expf((float)(2 * i) * (-9.210340371976184f / 512.0f));
  const float ang = (float)t * div;
  return (d & 1) ? __cosf(ang) : __sinf(ang);
}

// LayerNorm over 512; flags: 1 = add PE after, 2 = ReLU after; Res added before stats
__global__ void k_ln512(const float* __restrict__ X, const float* __restrict__ Res,
                        const float* __restrict__ g, const float* __restrict__ bt,
                        float* __restrict__ Y, int flags)
{
  __shared__ float sm[256];
  const long long row = blockIdx.x;
  const int tid = threadIdx.x;
  const float* xr = X + row * 512;
  float v0 = xr[tid], v1 = xr[tid + 256];
  if (Res) { const float* rr = Res + row * 512; v0 += rr[tid]; v1 += rr[tid + 256]; }
  sm[tid] = v0 + v1; __syncthreads();
  for (int s = 128; s > 0; s >>= 1) { if (tid < s) sm[tid] += sm[tid + s]; __syncthreads(); }
  const float mean = sm[0] * (1.f / 512.f); __syncthreads();
  const float d0 = v0 - mean, d1 = v1 - mean;
  sm[tid] = d0 * d0 + d1 * d1; __syncthreads();
  for (int s = 128; s > 0; s >>= 1) { if (tid < s) sm[tid] += sm[tid + s]; __syncthreads(); }
  const float inv = rsqrtf(sm[0] * (1.f / 512.f) + 1e-5f);
  const int t = (int)(row & 63);
  float y0 = d0 * inv * g[tid] + bt[tid];
  float y1 = d1 * inv * g[tid + 256] + bt[tid + 256];
  if (flags & 1) { y0 += pe_val(t, tid); y1 += pe_val(t, tid + 256); }
  if (flags & 2) { y0 = fmaxf(y0, 0.f); y1 = fmaxf(y1, 0.f); }
  Y[row * 512 + tid] = y0; Y[row * 512 + tid + 256] = y1;
}

// in-place row softmax over 64 (transformer attention)
__global__ void k_rowsoftmax(float* __restrict__ X)
{
  __shared__ float sm[64];
  const long long row = blockIdx.x;
  const int tid = threadIdx.x;
  const float v = X[(row << 6) + tid];
  sm[tid] = v; __syncthreads();
  for (int s = 32; s > 0; s >>= 1) { if (tid < s) sm[tid] = fmaxf(sm[tid], sm[tid + s]); __syncthreads(); }
  const float mx = sm[0]; __syncthreads();
  const float e = __expf(v - mx);
  sm[tid] = e; __syncthreads();
  for (int s = 32; s > 0; s >>= 1) { if (tid < s) sm[tid] += sm[tid + s]; __syncthreads(); }
  X[(row << 6) + tid] = e / sm[0];
}

// c1/c3/c5 single-channel 3D convs over (15,64,64) -> cat[b][45][4096]
__global__ void k_c135(const float* __restrict__ xm, const float* __restrict__ w1,
                       const float* __restrict__ w3, const float* __restrict__ w5,
                       float* __restrict__ cat)
{
  const int idx = blockIdx.x * 256 + threadIdx.x;
  if (idx >= NB * 45 * 4096) return;
  const int p = idx & 4095, cc = (idx >> 12) % 45, b = idx / (45 * 4096);
  const int x = p & 63, y = p >> 6;
  float acc = 0.f;
  if (cc < 15) {
    acc = w1[0] * xm[((long long)b * 15 + cc) * 4096 + p];
  } else if (cc < 30) {
    const int c = cc - 15;
    for (int dc = 0; dc < 3; ++dc) for (int dy = 0; dy < 3; ++dy) for (int dx = 0; dx < 3; ++dx) {
      const int ci = c + dc - 1, yy = y + dy - 1, xx = x + dx - 1;
      if ((unsigned)ci < 15u && (unsigned)yy < 64u && (unsigned)xx < 64u)
        acc += w3[(dc * 3 + dy) * 3 + dx] * xm[((long long)b * 15 + ci) * 4096 + yy * 64 + xx];
    }
  } else {
    const int c = cc - 30;
    for (int dc = 0; dc < 5; ++dc) for (int dy = 0; dy < 5; ++dy) for (int dx = 0; dx < 5; ++dx) {
      const int ci = c + dc - 2, yy = y + dy - 2, xx = x + dx - 2;
      if ((unsigned)ci < 15u && (unsigned)yy < 64u && (unsigned)xx < 64u)
        acc += w5[(dc * 5 + dy) * 5 + dx] * xm[((long long)b * 15 + ci) * 4096 + yy * 64 + xx];
    }
  }
  cat[idx] = acc;
}

// allc[b][d] = sum_{kd,p} call_w[kd*4096+p] * cat[b][3d+kd][p]  (block reduce)
__global__ void k_call(const float* __restrict__ cat, const float* __restrict__ cw,
                       float* __restrict__ allc)
{
  __shared__ float sm[256];
  const int b = blockIdx.x / 15, d = blockIdx.x % 15;
  float acc = 0.f;
  for (int i = threadIdx.x; i < 3 * 4096; i += 256) {
    const int kd = i >> 12, p = i & 4095;
    acc += cw[kd * 4096 + p] * cat[((long long)b * 45 + 3 * d + kd) * 4096 + p];
  }
  sm[threadIdx.x] = acc; __syncthreads();
  for (int s = 128; s > 0; s >>= 1) { if (threadIdx.x < s) sm[threadIdx.x] += sm[threadIdx.x + s]; __syncthreads(); }
  if (threadIdx.x == 0) allc[b * 15 + d] = sm[0];
}

__global__ void k_add15(const float* __restrict__ xm, const float* __restrict__ allc,
                        float* __restrict__ x15)
{
  const int idx = blockIdx.x * 256 + threadIdx.x;
  if (idx >= NB * 15 * 4096) return;
  const int c = (idx >> 12) % 15, b = idx / (15 * 4096);
  x15[idx] = xm[idx] + allc[b * 15 + c];
}

// conv2 15->32 3x3 pad1 + BN2 + ReLU, NHWC output [b][y][x][32]
__global__ void k_conv2(const float* __restrict__ x15, const float* __restrict__ w,
                        const float* __restrict__ bb, const float* __restrict__ g,
                        const float* __restrict__ bt, const float* __restrict__ mean,
                        const float* __restrict__ var, float* __restrict__ xt)
{
  const int idx = blockIdx.x * 256 + threadIdx.x;
  if (idx >= NB * 64 * 64 * 32) return;
  const int c = idx & 31, x = (idx >> 5) & 63, y = (idx >> 11) & 63, b = idx >> 17;
  float acc = bb[c];
  for (int ci = 0; ci < 15; ++ci)
    for (int dy = 0; dy < 3; ++dy) {
      const int yy = y + dy - 1; if ((unsigned)yy >= 64u) continue;
      for (int dx = 0; dx < 3; ++dx) {
        const int xx = x + dx - 1; if ((unsigned)xx >= 64u) continue;
        acc += w[((c * 15 + ci) * 3 + dy) * 3 + dx] *
               x15[((long long)b * 15 + ci) * 4096 + yy * 64 + xx];
      }
    }
  const float sc = g[c] * rsqrtf(var[c] + 1e-5f);
  xt[idx] = fmaxf((acc - mean[c]) * sc + bt[c], 0.f);
}

// final 256 -> 1 dot
__global__ void k_pred3(const float* __restrict__ p2, const float* __restrict__ w,
                        const float* __restrict__ b, float* __restrict__ h)
{
  const int r = threadIdx.x;
  if (r >= 256) return;
  float acc = b[0];
  for (int i = 0; i < 256; ++i) acc += p2[r * 256 + i] * w[i];
  h[r] = acc;
}

// ===========================================================================
// Host orchestration
// ===========================================================================
static inline void gemm_bt(hipStream_t st, const float* A, const float* B, const float* bias,
                           const float* Res, float* C, int M, int N, int K,
                           int lda, int ldb, int ldc, int flags = 0, float alpha = 1.f,
                           int gz = 1, int zdiv = 1,
                           long long aZ1 = 0, long long aZ2 = 0,
                           long long bZ1 = 0, long long bZ2 = 0,
                           long long cZ1 = 0, long long cZ2 = 0)
{
  dim3 g((M + 63) / 64, (N + 31) / 32, gz);
  k_gemm<true><<<g, 256, 0, st>>>(A, B, bias, Res, C, M, N, K, lda, ldb, ldc,
                                  zdiv, aZ1, aZ2, bZ1, bZ2, cZ1, cZ2, alpha, flags);
}

static inline void gemm_nt(hipStream_t st, const float* A, const float* B, const float* bias,
                           const float* Res, float* C, int M, int N, int K,
                           int lda, int ldb, int ldc, int flags = 0, float alpha = 1.f,
                           int gz = 1, int zdiv = 1,
                           long long aZ1 = 0, long long aZ2 = 0,
                           long long bZ1 = 0, long long bZ2 = 0,
                           long long cZ1 = 0, long long cZ2 = 0)
{
  dim3 g((M + 63) / 64, (N + 31) / 32, gz);
  k_gemm<false><<<g, 256, 0, st>>>(A, B, bias, Res, C, M, N, K, lda, ldb, ldc,
                                   zdiv, aZ1, aZ2, bZ1, bZ2, cZ1, cZ2, alpha, flags);
}

extern "C" void kernel_launch(void* const* d_in, const int* in_sizes, int n_in,
                              void* d_out, int out_size, void* d_ws, size_t ws_size,
                              hipStream_t stream)
{
  (void)in_sizes; (void)n_in; (void)out_size; (void)ws_size;
#define P(i) ((const float*)d_in[(i)])
  const float* feats = P(IN_FEATS);
  float* out = (float*)d_out;
  float* ws  = (float*)d_ws;

  float* convout = ws + WS_CONVOUT;
  float* hbuf = ws + WS_H;   float* xs = ws + WS_XS;
  float* qb = ws + WS_Q;     float* kb = ws + WS_K;
  float* raw = ws + WS_RAW;  float* t1 = ws + WS_T1;
  float* tpw = ws + WS_TPW;  float* hatf = ws + WS_HATF;
  float* cat = ws + WS_CAT;  float* allc = ws + WS_ALLC;
  float* x15 = ws + WS_X15;  float* xt = ws + WS_XT;
  float* xp = ws + WS_XP;    float* x0 = ws + WS_X0;
  float* qkv = ws + WS_QKV;  float* raw2 = ws + WS_RAW2;
  float* obuf = ws + WS_OBUF;
  float* x1t = ws + WS_X1T;  float* x1 = ws + WS_X1;
  float* f1 = ws + WS_F1;
  float* x2t = ws + WS_X2T;  float* x2 = ws + WS_X2;
  float* p1t = ws + WS_P1T;  float* p1 = ws + WS_P1;
  float* p2 = ws + WS_P2;

  float* xmat = out + OUT_XMAT;
  float* m1o  = out + OUT_M1;
  float* m2o  = out + OUT_M2;

  // ---------------- per-scale tower ----------------
  for (int s = 0; s < 3; ++s) {
    const float* fs = feats + (long long)s * FEAT_PER_SCALE;
    // dilated conv3d as implicit GEMM on the matrix pipe
    k_conv3d<<<dim3(CONV_M / 64, CONV_N / 32, 1), 256, 0, stream>>>(fs, P(IN_CONV3D_W), convout);
    // bias + BN1 + ReLU + max over 7x7
    k_bnmax<<<(NB * ND * NT + 255) / 256, 256, 0, stream>>>(
        convout, P(IN_CONV3D_B), P(IN_BN1_G), P(IN_BN1_B), P(IN_BN1_M), P(IN_BN1_V), hbuf, xs);
    // q / k projections
    gemm_bt(stream, xs, P(IN_WQ), P(IN_BQ), nullptr, qb, 256, 512, 512, 512, 512, 512);
    gemm_bt(stream, xs, P(IN_WK), P(IN_BK), nullptr, kb, 256, 512, 512, 512, 512, 512);
    // raw[b][h] = q_h @ k_h^T (batched over z = b*4+h)
    gemm_bt(stream, qb, kb, nullptr, nullptr, raw, 64, 64, 128, 512, 512, 64,
            0, 1.f, 16, 4, 32768, 128, 32768, 128, 16384, 4096);
    k_attn_soft<<<16, 64, 0, stream>>>(raw, xmat, m1o, m2o, s);
    // hat branch: depthwise -> pointwise -> fc -> LN64
    k_dwconv<<<(NB * ND * NT + 255) / 256, 256, 0, stream>>>(hbuf, P(IN_DW_W), P(IN_DW_B), t1);
    gemm_bt(stream, t1, P(IN_PW_W), P(IN_PW_B), nullptr, tpw, 256, 512, 512, 512, 512, 512);
    gemm_bt(stream, tpw, P(IN_FC_W), P(IN_FC_B), nullptr, hatf, 256, 64, 512, 512, 512, 64);
    k_ln64_hat<<<256, 64, 0, stream>>>(hatf, P(IN_FCLN_G), P(IN_FCLN_B), xmat, s);
  }

  // ---------------- multi-scale fusion convs ----------------
  k_c135<<<(NB * 45 * 4096 + 255) / 256, 256, 0, stream>>>(xmat, P(IN_C1_W), P(IN_C3_W), P(IN_C5_W), cat);
  k_call<<<60, 256, 0, stream>>>(cat, P(IN_CALL_W), allc);
  k_add15<<<(NB * 15 * 4096 + 255) / 256, 256, 0, stream>>>(xmat, allc, x15);
  k_conv2<<<(NB * 64 * 64 * 32 + 255) / 256, 256, 0, stream>>>(
      x15, P(IN_CONV2_W), P(IN_CONV2_B), P(IN_BN2_G), P(IN_BN2_B), P(IN_BN2_M), P(IN_BN2_V), xt);

  // proj (2048 -> 512) + ReLU, LN1, +PE
  gemm_bt(stream, xt, P(IN_PROJ_W), P(IN_PROJ_B), nullptr, xp, 256, 512, 2048, 2048, 2048, 512, 1);
  k_ln512<<<256, 256, 0, stream>>>(xp, nullptr, P(IN_LN1_G), P(IN_LN1_B), x0, 1);

  // ---------------- transformer encoder ----------------
  gemm_bt(stream, x0, P(IN_ATTN_IN_W), P(IN_ATTN_IN_B), nullptr, qkv, 256, 1536, 512, 512, 512, 1536);
  // logits = (q_h @ k_h^T) / sqrt(128), batched over (b,h)
  gemm_bt(stream, qkv, qkv + 512, nullptr, nullptr, raw2, 64, 64, 128, 1536, 1536, 64,
          0, 0.08838834764831845f, 16, 4, 98304, 128, 98304, 128, 16384, 4096);
  k_rowsoftmax<<<1024, 64, 0, stream>>>(raw2);
  // o_h = att_h @ v_h
  gemm_nt(stream, raw2, qkv + 1024, nullptr, nullptr, obuf, 64, 128, 64, 64, 1536, 512,
          0, 1.f, 16, 4, 16384, 4096, 98304, 128, 32768, 128);
  gemm_bt(stream, obuf, P(IN_ATTN_OUT_W), P(IN_ATTN_OUT_B), x0, x1t, 256, 512, 512, 512, 512, 512);
  k_ln512<<<256, 256, 0, stream>>>(x1t, nullptr, P(IN_ELN1_G), P(IN_ELN1_B), x1, 0);
  gemm_bt(stream, x1, P(IN_FF1_W), P(IN_FF1_B), nullptr, f1, 256, 512, 512, 512, 512, 512, 1);
  gemm_bt(stream, f1, P(IN_FF2_W), P(IN_FF2_B), x1, x2t, 256, 512, 512, 512, 512, 512);
  k_ln512<<<256, 256, 0, stream>>>(x2t, nullptr, P(IN_ELN2_G), P(IN_ELN2_B), x2, 0);
  k_ln512<<<256, 256, 0, stream>>>(x2, nullptr, P(IN_ENORM_G), P(IN_ENORM_B), out + OUT_O, 0);

  // ---------------- prediction head ----------------
  gemm_bt(stream, out + OUT_O, P(IN_PRED1_W), P(IN_PRED1_B), nullptr, p1t, 256, 512, 512, 512, 512, 512);
  k_ln512<<<256, 256, 0, stream>>>(p1t, nullptr, P(IN_PREDLN_G), P(IN_PREDLN_B), p1, 2);
  gemm_bt(stream, p1, P(IN_PRED2_W), P(IN_PRED2_B), nullptr, p2, 256, 256, 512, 512, 512, 256, 1);
  k_pred3<<<1, 256, 0, stream>>>(p2, P(IN_PRED3_W), P(IN_PRED3_B), out + OUT_H);
#undef P
}
